// Base_90134183674571
// MI455X (gfx1250) — matrix operand order repaired and checked by reading, not compiled, not run on hardware
//
#include <hip/hip_runtime.h>
#include <hip/hip_bf16.h>

// Problem dimensions (fixed by the reference setup_inputs()).
constexpr int B  = 32;
constexpr int C  = 192;
constexpr int Tx = 256;
constexpr int Ty = 1024;
constexpr int K2 = 2 * C;           // fused GEMM K dimension = 384
constexpr float NEG    = -1e9f;
constexpr float LOG2PI = 1.8378770664093453f;

typedef __attribute__((ext_vector_type(16))) __bf16 v16bf;
typedef __attribute__((ext_vector_type(8)))  float  v8f;

// ---------------- workspace layout (bytes) ----------------
// A   : [B][Tx][K2] bf16   (o_scale | m_p*o_scale)          6,291,456
// Bm  : [B][Ty][K2] bf16   (-0.5 z^2 | z)                  25,165,824
// bias: [B][Tx]     f32    (logp1 + logp4)                     32,768
// lp  : [B][Ty][Tx] f32    masked logp (read-only after GEMM) 33,554,432
// dir : [B][Ty][32] u8     backtrack decision bits            1,048,576
constexpr size_t OFF_A    = 0;
constexpr size_t OFF_B    = OFF_A + (size_t)B * Tx * K2 * 2;
constexpr size_t OFF_BIAS = OFF_B + (size_t)B * Ty * K2 * 2;
constexpr size_t OFF_LP   = OFF_BIAS + (size_t)B * Tx * 4;
constexpr size_t OFF_DIR  = OFF_LP + (size_t)B * Ty * Tx * 4;

// ------------- prep A operand + bias  (one block per (b,t)) -------------
__global__ void __launch_bounds__(256) prep_A(const float* __restrict__ m_p,
                                              const float* __restrict__ logs_p,
                                              __bf16* __restrict__ A,
                                              float* __restrict__ bias) {
  const int blk = blockIdx.x;
  const int b = blk / Tx, t = blk % Tx;
  const int c = threadIdx.x;                 // 256 threads, only c<192 active
  __shared__ float red[256];
  float partial = 0.0f;
  if (c < C) {
    const size_t src = ((size_t)b * C + c) * Tx + t;
    const float ls = logs_p[src];
    const float m  = m_p[src];
    const float o  = expf(-2.0f * ls);       // o_scale
    __bf16* row = A + ((size_t)b * Tx + t) * K2;
    row[c]     = (__bf16)o;                  // multiplies (-0.5 z^2)  -> logp2
    row[C + c] = (__bf16)(m * o);            // multiplies z           -> logp3
    partial = -0.5f * LOG2PI - ls - 0.5f * m * m * o;   // logp1 + logp4 terms
  }
  red[c] = partial;
  __syncthreads();
  for (int s = 128; s > 0; s >>= 1) {
    if (c < s) red[c] += red[c + s];
    __syncthreads();
  }
  if (c == 0) bias[(size_t)b * Tx + t] = red[0];
}

// ------------- prep B operand  (one block per (b,s)) -------------
__global__ void __launch_bounds__(256) prep_B(const float* __restrict__ z_p,
                                              __bf16* __restrict__ Bm) {
  const int blk = blockIdx.x;
  const int b = blk / Ty, s = blk % Ty;
  const int c = threadIdx.x;
  if (c < C) {
    const float z = z_p[((size_t)b * C + c) * Ty + s];
    __bf16* row = Bm + ((size_t)b * Ty + s) * K2;
    row[c]     = (__bf16)(-0.5f * z * z);
    row[C + c] = (__bf16)z;
  }
}

// ------------- fused WMMA GEMM: lp[b][s][t] = (A@B + bias) * mask -------------
// One wave computes a 16(M=Tx) x 64(N=Ty) strip = 4 accumulators, reusing the
// A fragment 4x. K-loop: 384/32 = 12 steps of v_wmma_f32_16x16x32_bf16.
__global__ void __launch_bounds__(256) wmma_logp(const __bf16* __restrict__ A,
                                                 const __bf16* __restrict__ Bm,
                                                 const float* __restrict__ bias,
                                                 const float* __restrict__ x_mask,
                                                 const float* __restrict__ y_mask,
                                                 float* __restrict__ lp) {
  const int wave   = threadIdx.x >> 5;
  const int lane   = threadIdx.x & 31;
  const int lane16 = lane & 15;
  const int hi     = lane >> 4;
  const int w  = blockIdx.x * 8 + wave;      // 8192 waves total
  const int b  = w >> 8;                     // 256 strips per batch
  const int r  = w & 255;
  const int t0 = (r >> 4) * 16;              // M tile origin (Tx)
  const int s0 = (r & 15) * 64;              // N strip origin (Ty)

  v8f acc[4] = {v8f{}, v8f{}, v8f{}, v8f{}};

  // A fragment per ISA 7.12.2 (16-bit A 16x32): lanes 0-15 carry K[0..7],[16..23],
  // lanes 16-31 carry K[8..15],[24..31]; row M = lane16.
  const __bf16* Arow  = A  + ((size_t)b * Tx + t0 + lane16) * K2 + hi * 8;
  // B fragment (16-bit B 32x16): lanes 0-15 K[0..7],[16..23], lanes 16-31 K[8..15],[24..31]; N = lane16.
  const __bf16* Brow0 = Bm + ((size_t)b * Ty + s0 + lane16) * K2 + hi * 8;

  for (int k0 = 0; k0 < K2; k0 += 32) {
    union { uint4 q[2]; v16bf v; } af;
    af.q[0] = *(const uint4*)(Arow + k0);
    af.q[1] = *(const uint4*)(Arow + k0 + 16);
#pragma unroll
    for (int j = 0; j < 4; ++j) {
      union { uint4 q[2]; v16bf v; } bfr;
      const __bf16* Br = Brow0 + (size_t)j * 16 * K2 + k0;
      bfr.q[0] = *(const uint4*)(Br);
      bfr.q[1] = *(const uint4*)(Br + 16);
      acc[j] = __builtin_amdgcn_wmma_f32_16x16x32_bf16(
          false, af.v, false, bfr.v, (short)0, acc[j], false, false);
    }
  }

  // Epilogue: C/D layout — VGPR e, lane<16 -> M=e, lane>=16 -> M=e+8; N=lane16.
  const int rowbase = t0 + hi * 8;
  float biasr[8], xmr[8];
#pragma unroll
  for (int e = 0; e < 8; ++e) {
    biasr[e] = bias[(size_t)b * Tx + rowbase + e];
    xmr[e]   = x_mask[(size_t)b * Tx + rowbase + e];
  }
#pragma unroll
  for (int j = 0; j < 4; ++j) {
    const int col = s0 + j * 16 + lane16;
    const float ym = y_mask[(size_t)b * Ty + col];
    float o[8];
#pragma unroll
    for (int e = 0; e < 8; ++e)
      o[e] = (acc[j][e] + biasr[e]) * (xmr[e] * ym);   // logp * mask
    float* dst = lp + ((size_t)b * Ty + col) * Tx + rowbase;
    *(float4*)(dst)     = make_float4(o[0], o[1], o[2], o[3]);
    *(float4*)(dst + 4) = make_float4(o[4], o[5], o[6], o[7]);
  }
}

// ------------- forward DP: ONE wave32 per batch, column in registers -------------
// Lane L owns t = L*8 .. L*8+7. The t-1 shift is 7 register moves + one
// __shfl_up across lanes — no barriers, no LDS, no values[] writeback.
// Instead of storing the 32 MB values array, emit the backtrack decision bit
// d[y][t] = (values[y-1][t-1] >= values[y-1][t]) — exactly the comparison the
// max() already performs — packed 8 bits/lane -> 32 bytes per (b,y).
__global__ void __launch_bounds__(32) dp_forward(const float* __restrict__ lp,
                                                 unsigned char* __restrict__ dir) {
  constexpr int PF = 4;                       // row prefetch depth
  const int b = blockIdx.x;
  const int lane = threadIdx.x;
  const float* base = lp + (size_t)b * Ty * Tx + lane * 8;
  unsigned char* dbase = dir + (size_t)b * Ty * 32 + lane;

  float prev[8];
#pragma unroll
  for (int i = 0; i < 8; ++i) prev[i] = NEG;
  if (lane == 0) prev[0] = 0.0f;              // init column: t==0 -> 0

  float4 buf[PF][2];
#pragma unroll
  for (int p = 0; p < PF; ++p) {
    buf[p][0] = *(const float4*)(base + (size_t)p * Tx);
    buf[p][1] = *(const float4*)(base + (size_t)p * Tx + 4);
  }

  for (int yb = 0; yb < Ty; yb += PF) {
#pragma unroll
    for (int u = 0; u < PF; ++u) {
      const int y = yb + u;
      float row[8] = {buf[u][0].x, buf[u][0].y, buf[u][0].z, buf[u][0].w,
                      buf[u][1].x, buf[u][1].y, buf[u][1].z, buf[u][1].w};
      if (y + PF < Ty) {                      // refill slot PF rows ahead
        buf[u][0] = *(const float4*)(base + (size_t)(y + PF) * Tx);
        buf[u][1] = *(const float4*)(base + (size_t)(y + PF) * Tx + 4);
      }
      const float up = __shfl_up(prev[7], 1, 32);
      float sh[8];
      sh[0] = (lane == 0) ? NEG : up;
#pragma unroll
      for (int i = 1; i < 8; ++i) sh[i] = prev[i - 1];
      unsigned int bits = 0;
#pragma unroll
      for (int i = 0; i < 8; ++i) {
        bits |= (sh[i] >= prev[i]) ? (1u << i) : 0u;   // v_move >= v_stay
        prev[i] = row[i] + fmaxf(prev[i], sh[i]);      // cur = logp + max(...)
      }
      dbase[(size_t)y * 32] = (unsigned char)bits;
    }
  }
}

// ------------- zero the output -------------
__global__ void zero_out(float4* __restrict__ out, int n4) {
  const int i = blockIdx.x * blockDim.x + threadIdx.x;
  if (i < n4) out[i] = make_float4(0.f, 0.f, 0.f, 0.f);
}

// ------------- backtrack: one block per batch, decision bits in LDS -------------
__global__ void __launch_bounds__(256) backtrack(const unsigned char* __restrict__ dir,
                                                 const float* __restrict__ x_mask,
                                                 const float* __restrict__ y_mask,
                                                 float* __restrict__ out) {
  const int b = blockIdx.x;
  __shared__ unsigned char sdir[Ty * 32];     // 32 KB bitmap for this batch
  {
    const uint4* src = (const uint4*)(dir + (size_t)b * Ty * 32);
    uint4* dst = (uint4*)sdir;
    for (int i = threadIdx.x; i < Ty * 32 / 16; i += 256) dst[i] = src[i];
  }
  __syncthreads();
  if (threadIdx.x != 0) return;

  int tx = 0, ty = 0;
  for (int t = 0; t < Tx; ++t) tx += (x_mask[(size_t)b * Tx + t] != 0.0f);
  for (int s = 0; s < Ty; ++s) ty += (y_mask[(size_t)b * Ty + s] != 0.0f);

  int index = tx - 1;
  float* ob = out + (size_t)b * Tx * Ty;
  for (int y = Ty - 1; y >= 0; --y) {
    const bool active = y < ty;
    if (active && index >= 0)
      ob[(size_t)index * Ty + y] = 1.0f;
    const int ci = index < 0 ? 0 : index;
    const int bit = (sdir[y * 32 + (ci >> 3)] >> (ci & 7)) & 1;
    const bool move = (index != 0) && ((index == y) || bit) && active && (y > 0);
    if (move) --index;
  }
}

extern "C" void kernel_launch(void* const* d_in, const int* in_sizes, int n_in,
                              void* d_out, int out_size, void* d_ws, size_t ws_size,
                              hipStream_t stream) {
  const float* z_p    = (const float*)d_in[0];
  const float* m_p    = (const float*)d_in[1];
  const float* logs_p = (const float*)d_in[2];
  const float* x_mask = (const float*)d_in[3];
  const float* y_mask = (const float*)d_in[4];
  float* out = (float*)d_out;

  char* ws = (char*)d_ws;
  __bf16* A    = (__bf16*)(ws + OFF_A);
  __bf16* Bm   = (__bf16*)(ws + OFF_B);
  float*  bias = (float*)(ws + OFF_BIAS);
  float*  lp   = (float*)(ws + OFF_LP);
  unsigned char* dir = (unsigned char*)(ws + OFF_DIR);

  // 1) operand prep (bf16 conversion + bias reduction)
  prep_A<<<B * Tx, 256, 0, stream>>>(m_p, logs_p, A, bias);
  prep_B<<<B * Ty, 256, 0, stream>>>(z_p, Bm);

  // 2) fused bf16 WMMA GEMM -> masked logp [B,Ty,Tx]
  wmma_logp<<<1024, 256, 0, stream>>>(A, Bm, bias, x_mask, y_mask, lp);

  // 3) forward DP, one wave32 per batch, emits 1 MB of decision bits
  dp_forward<<<B, 32, 0, stream>>>(lp, dir);

  // 4) zero output, then LDS-resident backtrack of the hard path
  const int n4 = out_size / 4;                // out_size = 8,388,608 floats
  zero_out<<<(n4 + 255) / 256, 256, 0, stream>>>((float4*)out, n4);
  backtrack<<<B, 256, 0, stream>>>(dir, x_mask, y_mask, out);
}
